// TKConv2dR_71605694758981
// MI455X (gfx1250) — compile-verified, matrix operand order checked by reading, not run on hardware
//
#include <hip/hip_runtime.h>

// Tucker-factored conv2d for MI455X (gfx1250), bf16 WMMA path, channel-last intermediates.
//   stage1: y[b,p,s]  = sum_c first[s,c] * x[b,c,p]              (GEMM 64x256xN, x staged/transposed via LDS)
//   stage2: z[b,p,r]  = sum_{s,kh,kw} core[r,s,kh,kw]*y[b,p',s]  (implicit GEMM K=576, B frags direct from global)
//   stage3: out[b,o,p]= sum_r last[o,r] * z[b,p,r] + bias[o]     (GEMM 256x64xN, B frags direct from global)
// 256-thread block = 8 waves; block tile 64(M)x64(N); wave tile 16x32 (2 accumulators);
// K chunked by 32; v_wmma_f32_16x16x32_bf16, fp32 accumulate.

typedef __attribute__((ext_vector_type(16))) __bf16 v16bf;
typedef __attribute__((ext_vector_type(8)))  __bf16 v8bf;
typedef __attribute__((ext_vector_type(8)))  float  v8f;

#define HW    3136   // 56*56
#define W2    56
#define NB    32
#define CIN   256
#define COUT  256
#define RANK  64

#define A1_STR 264   // stage1 A LDS row stride (bf16): 528B rows -> conflict-free b128 frag reads
#define A2_STR 200   // stage2 A LDS row stride (per kh-group of 192 K-values + pad)
#define A3_STR 72    // stage3 A LDS row stride
#define BS_STR 40    // stage1 B staging row stride (80B, 16B aligned)

// 16-bit WMMA operand layout (ISA 7.12.2): lanes 0-15 -> K = kk..kk+7 & 16+kk..,
// with kk=0; lanes 16-31 -> kk=8. Fragment = two 16B loads.
__device__ __forceinline__ v16bf frag_cat(v8bf lo, v8bf hi) {
  v16bf r;
#pragma unroll
  for (int i = 0; i < 8; ++i) { r[i] = lo[i]; r[i + 8] = hi[i]; }
  return r;
}
__device__ __forceinline__ v16bf frag16(const __bf16* p) {   // LDS or global, 16B aligned
  return frag_cat(*(const v8bf*)p, *(const v8bf*)(p + 16));
}
__device__ __forceinline__ unsigned pack2(float a, float b) { // -> v_cvt_pk_bf16_f32
  union { unsigned u; __bf16 h[2]; } k;
  k.h[0] = (__bf16)a; k.h[1] = (__bf16)b; return k.u;
}
__device__ __forceinline__ void store_d8(__bf16* dst, v8f acc) { // packed b128 D store
  v8bf o;
#pragma unroll
  for (int i = 0; i < 8; ++i) o[i] = (__bf16)acc[i];
  *(v8bf*)dst = o;
}
__device__ __forceinline__ v8f wmma_bf16(v16bf a, v16bf b, v8f c) {
  return __builtin_amdgcn_wmma_f32_16x16x32_bf16(false, a, false, b, (short)0, c, false, false);
}

// ---------------------------------------------------------------- stage 1
__global__ __launch_bounds__(256)
void tk_stage1(const float* __restrict__ x, const float* __restrict__ first,
               __bf16* __restrict__ y) {
  __shared__ __bf16 lds_a[RANK * A1_STR];   // all of first, bf16
  __shared__ __bf16 lds_b[64 * BS_STR];     // x chunk, pixel-major
  const int t = threadIdx.x, lane = t & 31, wave = t >> 5;
  const int mt = wave >> 1, nt = wave & 1;
  const int p0 = blockIdx.x * 64, b = blockIdx.y;
  const float* xb = x + (size_t)b * CIN * HW;

  for (int i = 0; i < 64; ++i) {            // preload A: first[64][256] -> bf16, once
    int idx = t + i * 256;                  // 16384 elems
    int m = idx >> 8, k = idx & 255;
    lds_a[m * A1_STR + k] = (__bf16)first[idx];
  }
  __syncthreads();

  const int kk  = (lane < 16) ? 0 : 8;
  const int ar  = mt * 16 + (lane & 15);
  const int bc  = nt * 32 + (lane & 15);
  v8f acc0 = {}, acc1 = {};
  for (int k0 = 0; k0 < CIN; k0 += 32) {
#pragma unroll
    for (int i = 0; i < 4; ++i) {           // stage x tile transposed, 2 channels/b32
      int idx = t + i * 256;                // 1024 dwords
      int c2 = idx >> 6, pl = idx & 63;
      float f0 = xb[(size_t)(k0 + 2 * c2    ) * HW + p0 + pl];
      float f1 = xb[(size_t)(k0 + 2 * c2 + 1) * HW + p0 + pl];
      *(unsigned*)&lds_b[pl * BS_STR + 2 * c2] = pack2(f0, f1);
    }
    __syncthreads();
    v16bf a  = frag16(&lds_a[ar * A1_STR + k0 + kk]);
    v16bf b0 = frag16(&lds_b[ bc       * BS_STR + kk]);
    v16bf b1 = frag16(&lds_b[(bc + 16) * BS_STR + kk]);
    acc0 = wmma_bf16(a, b0, acc0);
    acc1 = wmma_bf16(a, b1, acc1);
    __syncthreads();
  }
  const int row0 = mt * 16 + ((lane >> 4) << 3);  // 8 consecutive M rows per lane
  store_d8(y + ((size_t)b * HW + p0 + bc     ) * RANK + row0, acc0);
  store_d8(y + ((size_t)b * HW + p0 + bc + 16) * RANK + row0, acc1);
}

// ---------------------------------------------------------------- stage 2
__global__ __launch_bounds__(256)
void tk_stage2(const __bf16* __restrict__ y, const float* __restrict__ core,
               __bf16* __restrict__ z) {
  __shared__ __bf16 lds_a[RANK * A2_STR];   // core, one kh-row (192 K-values) at a time
  const int t = threadIdx.x, lane = t & 31, wave = t >> 5;
  const int mt = wave >> 1, nt = wave & 1;
  const int p0 = blockIdx.x * 64, b = blockIdx.y;
  const __bf16* yb = y + (size_t)b * HW * RANK;

  const int kk  = (lane < 16) ? 0 : 8;
  const int ar  = mt * 16 + (lane & 15);
  const int pb0 = p0 + nt * 32 + (lane & 15);
  const int pb1 = pb0 + 16;
  const int h0a = pb0 / W2, w0a = pb0 % W2;
  const int h0b = pb1 / W2, w0b = pb1 % W2;
  v8f acc0 = {}, acc1 = {};
  const v16bf vzero = {};

  for (int g = 0; g < 3; ++g) {             // kh groups
    __syncthreads();                        // previous group's frags consumed
    for (int i = 0; i < 48; ++i) {          // preload core[r][s][g][kw] -> K-packed
      int idx = t + i * 256;                // 12288 elems
      int r = idx / 192, km = idx % 192;
      int kc = km >> 5, k = km & 31;        // kc: (kw, s-half)
      int kwi = kc >> 1, s0 = (kc & 1) * 32;
      lds_a[r * A2_STR + km] = (__bf16)core[((r * RANK + s0 + k) * 3 + g) * 3 + kwi];
    }
    __syncthreads();
    const int dh = g - 1;
#pragma unroll
    for (int kc = 0; kc < 6; ++kc) {
      const int dw = (kc >> 1) - 1;
      const int s0 = (kc & 1) * 32;
      v16bf a = frag16(&lds_a[ar * A2_STR + kc * 32 + kk]);
      // B frags direct from global (channel-last y), zero halo via select
      int ha = h0a + dh, wa = w0a + dw;
      bool oka = ((unsigned)ha < 56u) && ((unsigned)wa < 56u);
      int pa = oka ? (ha * W2 + wa) : 0;
      v16bf b0 = frag16(yb + (size_t)pa * RANK + s0 + kk);
      b0 = oka ? b0 : vzero;
      int hb = h0b + dh, wb = w0b + dw;
      bool okb = ((unsigned)hb < 56u) && ((unsigned)wb < 56u);
      int pb = okb ? (hb * W2 + wb) : 0;
      v16bf b1 = frag16(yb + (size_t)pb * RANK + s0 + kk);
      b1 = okb ? b1 : vzero;
      acc0 = wmma_bf16(a, b0, acc0);
      acc1 = wmma_bf16(a, b1, acc1);
    }
  }
  const int row0 = mt * 16 + ((lane >> 4) << 3);
  store_d8(z + ((size_t)b * HW + pb0) * RANK + row0, acc0);
  store_d8(z + ((size_t)b * HW + pb1) * RANK + row0, acc1);
}

// ---------------------------------------------------------------- stage 3
__global__ __launch_bounds__(256)
void tk_stage3(const __bf16* __restrict__ z, const float* __restrict__ last,
               const float* __restrict__ bias, float* __restrict__ out) {
  __shared__ __bf16 lds_a[64 * A3_STR];
  const int t = threadIdx.x, lane = t & 31, wave = t >> 5;
  const int mt = wave >> 1, nt = wave & 1;
  const int p0 = blockIdx.x * 64;
  const int o0 = blockIdx.y * 64;
  const int b  = blockIdx.z;
  const __bf16* zb = z + (size_t)b * HW * RANK;

#pragma unroll
  for (int i = 0; i < 16; ++i) {            // preload last[o0..o0+63][0..63] -> bf16
    int idx = t + i * 256;                  // 4096 elems
    int ol = idx >> 6, r = idx & 63;
    lds_a[ol * A3_STR + r] = (__bf16)last[(o0 + ol) * RANK + r];
  }
  __syncthreads();

  const int kk  = (lane < 16) ? 0 : 8;
  const int ar  = mt * 16 + (lane & 15);
  const int pb0 = p0 + nt * 32 + (lane & 15);
  const int pb1 = pb0 + 16;
  v8f acc0 = {}, acc1 = {};
#pragma unroll
  for (int s0 = 0; s0 < RANK; s0 += 32) {
    v16bf a  = frag16(&lds_a[ar * A3_STR + s0 + kk]);
    v16bf b0 = frag16(zb + (size_t)pb0 * RANK + s0 + kk);
    v16bf b1 = frag16(zb + (size_t)pb1 * RANK + s0 + kk);
    acc0 = wmma_bf16(a, b0, acc0);
    acc1 = wmma_bf16(a, b1, acc1);
  }
  const int row0 = mt * 16 + ((lane >> 4) << 3);
#pragma unroll
  for (int v = 0; v < 8; ++v) {
    int o = o0 + row0 + v;
    float bo = bias[o];
    out[((size_t)b * COUT + o) * HW + pb0] = acc0[v] + bo;
    out[((size_t)b * COUT + o) * HW + pb1] = acc1[v] + bo;
  }
}

// ---------------------------------------------------------------- launcher
extern "C" void kernel_launch(void* const* d_in, const int* in_sizes, int n_in,
                              void* d_out, int out_size, void* d_ws, size_t ws_size,
                              hipStream_t stream) {
  const float* x     = (const float*)d_in[0];  // [32,256,56,56]
  const float* first = (const float*)d_in[1];  // [64,256]
  const float* core  = (const float*)d_in[2];  // [64,64,3,3]
  const float* last  = (const float*)d_in[3];  // [256,64]
  const float* bias  = (const float*)d_in[4];  // [256]
  float* out = (float*)d_out;                  // [32,256,56,56] fp32

  __bf16* y = (__bf16*)d_ws;                   // [32,3136,64] bf16 : 12.85 MB
  __bf16* z = y + (size_t)NB * HW * RANK;      // [32,3136,64] bf16 : 12.85 MB

  dim3 blk(256);
  tk_stage1<<<dim3(49, NB),    blk, 0, stream>>>(x, first, y);
  tk_stage2<<<dim3(49, NB),    blk, 0, stream>>>(y, core, z);
  tk_stage3<<<dim3(49, 4, NB), blk, 0, stream>>>(z, last, bias, out);
}
